// FCNet_74878459838520
// MI455X (gfx1250) — matrix operand order repaired and checked by reading, not compiled, not run on hardware
//
#include <hip/hip_runtime.h>

// ---------------------------------------------------------------------------
// Fused 8-layer MLP (patch reconstruction net) for gfx1250 / MI455X.
//
//   h = x @ W0^T + b0
//   repeat k = 0..6:  h = relu(h); h = h @ Wh[k]^T + bh[k]
//
// Strategy (compute-bound, AI ~1200 FLOP/B):
//   * weights converted once f32 -> bf16 into d_ws (fits in 192 MB L2)
//   * one workgroup owns a 64-row tile; the 64x1024 bf16 activation tile
//     lives in LDS for the entire network (no HBM traffic between layers)
//   * math: v_wmma_f32_16x16x32_bf16 (f32 accumulate)
//   * 1024 threads = 32 wave32 waves: 4 row-subtiles x 8 col-groups(128)
// ---------------------------------------------------------------------------

typedef __attribute__((ext_vector_type(16))) __bf16 v16bf;
typedef __attribute__((ext_vector_type(8)))  __bf16 v8bf;
typedef __attribute__((ext_vector_type(8)))  float  v8f;

#define VEC      1024      // hidden width
#define ROWS     8192      // B*P
#define MTILE    64        // rows per workgroup
#define PITCH    1032      // LDS row pitch in bf16 elems (1024 + 8 pad: 516 dw % 64 = 4)
#define NT       8         // 16-wide N tiles per wave (8*16 = 128 cols)
#define THREADS  1024      // 32 waves

// round-to-nearest-even f32 -> bf16 (bit pattern as ushort)
static __device__ __forceinline__ unsigned short f2bf(float f) {
    union { float f; unsigned int u; } v; v.f = f;
    unsigned int u = v.u;
    return (unsigned short)((u + 0x7FFFu + ((u >> 16) & 1u)) >> 16);
}

// A fragment: 16x32 bf16, per-ISA layout:
//   lanes 0-15  (M=lane):    V0-3 = K 0..7,  V4-7 = K 16..23
//   lanes 16-31 (M=lane-16): V0-3 = K 8..15, V4-7 = K 24..31
static __device__ __forceinline__ v16bf load_a(const unsigned short* hbuf,
                                               int msub, int kb, int lane) {
    const int row = msub * 16 + (lane & 15);
    const int hi  = (lane >> 4) & 1;
    const __bf16* p = (const __bf16*)(hbuf + row * PITCH + kb + hi * 8);
    v8bf lo = *(const v8bf*)(p);        // K = kb + hi*8      .. +7   (16B aligned)
    v8bf hh = *(const v8bf*)(p + 16);   // K = kb + 16 + hi*8 .. +7
    return __builtin_shufflevector(lo, hh, 0,1,2,3,4,5,6,7,8,9,10,11,12,13,14,15);
}

// B fragment: 32x16 bf16 (K x N). Column n of B = row n of W (row-major [o,i]),
// same K layout as the A fragment:
//   lanes 0-15  (N=lane):    V0-3 = K 0..7,  V4-7 = K 16..23
//   lanes 16-31 (N=lane-16): V0-3 = K 8..15, V4-7 = K 24..31
static __device__ __forceinline__ v16bf load_b(const unsigned short* __restrict__ W,
                                               int wstride, int col, int kb, int lane) {
    const int c  = col + (lane & 15);
    const int hi = (lane >> 4) & 1;
    const __bf16* p = (const __bf16*)(W + (size_t)c * wstride + kb + hi * 8);
    v8bf lo = *(const v8bf*)(p);
    v8bf hh = *(const v8bf*)(p + 16);
    return __builtin_shufflevector(lo, hh, 0,1,2,3,4,5,6,7,8,9,10,11,12,13,14,15);
}

// One dense layer over the LDS-resident activation tile.
//   reads  hbuf[0..63][0..KDIM)           (bf16)
//   writes hbuf (RELU path, bf16) or global out (final layer, f32)
template <int KDIM, bool RELU, bool TO_GLOBAL>
static __device__ __forceinline__ void mlp_layer(unsigned short* hbuf,
                                                 const unsigned short* __restrict__ W,
                                                 int wstride,
                                                 const float* __restrict__ bias,
                                                 float* __restrict__ outp,
                                                 int rowBase, int lane, int msub, int colbase) {
    v8f acc[NT];
#pragma unroll
    for (int t = 0; t < NT; ++t) acc[t] = {};

#pragma unroll 2
    for (int kb = 0; kb < KDIM; kb += 32) {
        const v16bf a = load_a(hbuf, msub, kb, lane);
#pragma unroll
        for (int t = 0; t < NT; ++t) {
            const v16bf b = load_b(W, wstride, colbase + t * 16, kb, lane);
            acc[t] = __builtin_amdgcn_wmma_f32_16x16x32_bf16(
                false, a, false, b, (short)0, acc[t], false, false);
        }
    }

    __syncthreads();  // all waves done READING hbuf before anyone overwrites it

    // C/D layout: VGPR j -> row (msub*16 + hi*8 + j), col = lane&15 within tile
    const int hi = (lane >> 4) & 1;
    const int rl = msub * 16 + hi * 8;
    const int cl = lane & 15;
#pragma unroll
    for (int t = 0; t < NT; ++t) {
        const int col = colbase + t * 16 + cl;
        const float bv = bias[col];
#pragma unroll
        for (int j = 0; j < 8; ++j) {
            float v = acc[t][j] + bv;
            if (RELU) v = v > 0.0f ? v : 0.0f;
            if (TO_GLOBAL) {
                outp[(size_t)(rowBase + rl + j) * VEC + col] = v;
            } else {
                hbuf[(rl + j) * PITCH + col] = f2bf(v);
            }
        }
    }
    if (!TO_GLOBAL) __syncthreads();  // writes visible before next layer reads
}

__global__ __launch_bounds__(THREADS)
void fcnet_fused(const float* __restrict__ x,
                 const unsigned short* __restrict__ w0bf,
                 const float* __restrict__ b0,
                 const unsigned short* __restrict__ whbf,
                 const float* __restrict__ bh,
                 float* __restrict__ out) {
    extern __shared__ unsigned short hbuf[];  // MTILE * PITCH bf16 (~129 KB)

    const int tid     = threadIdx.x;
    const int lane    = tid & 31;
    const int wave    = tid >> 5;
    const int msub    = wave & 3;          // 4 row-subtiles of 16
    const int colbase = (wave >> 2) * 128; // 8 col-groups of 128
    const int rowBase = blockIdx.x * MTILE;

    // Stage the 64x64 f32 x-tile into LDS as bf16 (4 elems / thread, float4 loads)
    {
        const int idx = tid * 4;              // 4096 elems total
        const int r = idx >> 6;
        const int c = idx & 63;
        const float4 v = *(const float4*)(x + (size_t)(rowBase + r) * 64 + c);
        unsigned short* dst = hbuf + r * PITCH + c;
        dst[0] = f2bf(v.x); dst[1] = f2bf(v.y);
        dst[2] = f2bf(v.z); dst[3] = f2bf(v.w);
    }
    __syncthreads();

    // layer 0: K=64, relu'd into LDS
    mlp_layer<64, true, false>(hbuf, w0bf, 64, b0, nullptr,
                               rowBase, lane, msub, colbase);
    // hidden layers 0..5: K=1024, relu'd into LDS
    for (int k = 0; k < 6; ++k) {
        mlp_layer<VEC, true, false>(hbuf, whbf + (size_t)k * VEC * VEC, VEC,
                                    bh + k * VEC, nullptr,
                                    rowBase, lane, msub, colbase);
    }
    // final layer: K=1024, f32 to global, no relu
    mlp_layer<VEC, false, true>(hbuf, whbf + (size_t)6 * VEC * VEC, VEC,
                                bh + 6 * VEC, out,
                                rowBase, lane, msub, colbase);
}

// f32 -> bf16 weight conversion (one-time prep into d_ws; weights then live in L2)
__global__ void cvt_f32_bf16(const float* __restrict__ src,
                             unsigned short* __restrict__ dst, int n) {
    const int i = (blockIdx.x * blockDim.x + threadIdx.x) * 4;
    if (i + 3 < n) {
        const float4 v = *(const float4*)(src + i);
        dst[i + 0] = f2bf(v.x); dst[i + 1] = f2bf(v.y);
        dst[i + 2] = f2bf(v.z); dst[i + 3] = f2bf(v.w);
    }
}

extern "C" void kernel_launch(void* const* d_in, const int* in_sizes, int n_in,
                              void* d_out, int out_size, void* d_ws, size_t ws_size,
                              hipStream_t stream) {
    (void)in_sizes; (void)n_in; (void)out_size; (void)ws_size;

    const float* x  = (const float*)d_in[0];   // [8192, 64]
    const float* W0 = (const float*)d_in[1];   // [1024, 64]
    const float* b0 = (const float*)d_in[2];   // [1024]
    const float* Wh = (const float*)d_in[3];   // [7, 1024, 1024]
    const float* bh = (const float*)d_in[4];   // [7, 1024]
    float* out = (float*)d_out;                // [8192, 1024]

    // bf16 weights in workspace: W0 (128 KB) then Wh (14 MB)
    unsigned short* w0bf = (unsigned short*)d_ws;
    unsigned short* whbf = w0bf + 1024 * 64;

    const int n0 = 1024 * 64;
    const int nh = 7 * 1024 * 1024;
    cvt_f32_bf16<<<(n0 / 4 + 255) / 256, 256, 0, stream>>>(W0, w0bf, n0);
    cvt_f32_bf16<<<(nh / 4 + 255) / 256, 256, 0, stream>>>(Wh, whbf, nh);

    const size_t shmem = (size_t)MTILE * PITCH * sizeof(unsigned short); // ~129 KB
    fcnet_fused<<<ROWS / MTILE, THREADS, shmem, stream>>>(x, w0bf, b0, whbf, bh, out);
}
